// NonLocal2d_44564580663307
// MI455X (gfx1250) — matrix operand order repaired and checked by reading, not compiled, not run on hardware
//
#include <hip/hip_runtime.h>

// Problem constants (match reference)
#define Bc 2
#define Cc 512
#define Pc 256
#define Hc 96
#define Wc 96
#define Nc (Hc * Wc)          // 9216
#define MTc (Nc / 64)         // 144 key tiles
#define QSCALE 0.0625f        // 1/sqrt(P) folded into Q projection

typedef __attribute__((ext_vector_type(16))) __bf16 bf16x16;
typedef __attribute__((ext_vector_type(8)))  __bf16 bf16x8;
typedef __attribute__((ext_vector_type(8)))  float  f32x8;
typedef __attribute__((ext_vector_type(4)))  float  f32x4;
typedef __attribute__((ext_vector_type(4)))  unsigned int u32x4;
typedef __attribute__((ext_vector_type(4)))  int    i32x4;

// ---------------------------------------------------------------------------
// gfx1250 async global->LDS copy (ASYNCcnt) with compile-safe fallback.
// Builtin signature: (global int4* src, local int4* dst, imm offset, imm cpol)
// ---------------------------------------------------------------------------
#if defined(__gfx1250__) && \
    __has_builtin(__builtin_amdgcn_global_load_async_to_lds_b128) && \
    __has_builtin(__builtin_amdgcn_s_wait_asynccnt)
#define USE_ASYNC 1
#else
#define USE_ASYNC 0
#endif

static __device__ inline void cp_lds_b128(__bf16* dst, const __bf16* src) {
#if USE_ASYNC
  __builtin_amdgcn_global_load_async_to_lds_b128(
      (__attribute__((address_space(1))) i32x4*)src,
      (__attribute__((address_space(3))) i32x4*)dst, 0, 0);
#else
  *(u32x4*)dst = *(const u32x4*)src;
#endif
}
static __device__ inline void cp_wait() {
#if USE_ASYNC
  __builtin_amdgcn_s_wait_asynccnt(0);
#endif
}

// ---- WMMA fragment loaders (CDNA5 16x16x32 bf16 layouts, ISA 7.12.2) ----
// A (16x32): lane holds row M=lane%16; elems j<8 -> K=(lane/16)*8+j,
//            j>=8 -> K=16+(lane/16)*8+(j-8).
static __device__ inline bf16x16 load_a_frag(const __bf16* base, int ld, int lane) {
  const __bf16* p = base + (lane & 15) * ld + ((lane >> 4) << 3);
  bf16x8 lo = *(const bf16x8*)p;
  bf16x8 hi = *(const bf16x8*)(p + 16);
  return __builtin_shufflevector(lo, hi, 0,1,2,3,4,5,6,7,8,9,10,11,12,13,14,15);
}
// B (32x16): same K order as A: elems j<8 -> K=(lane/16)*8+j, j>=8 -> K=16+(lane/16)*8+(j-8), N=lane%16.
// Source stored as B^T row-major [N][K] => two 16B reads per lane.
static __device__ inline bf16x16 load_b_frag(const __bf16* base, int ld, int lane) {
  const __bf16* p = base + (lane & 15) * ld + ((lane >> 4) << 3);
  bf16x8 lo = *(const bf16x8*)p;
  bf16x8 hi = *(const bf16x8*)(p + 16);
  return __builtin_shufflevector(lo, hi, 0,1,2,3,4,5,6,7,8,9,10,11,12,13,14,15);
}
static __device__ inline f32x8 wmma_bf16(bf16x16 a, bf16x16 b, f32x8 c) {
  return __builtin_amdgcn_wmma_f32_16x16x32_bf16(false, a, false, b, (short)0, c,
                                                 false, false);
}
// Build an A fragment directly from row-major fp32 global memory (for W).
static __device__ inline bf16x16 a_frag_from_f32(const float* p) {
  f32x4 v0 = *(const f32x4*)(p);
  f32x4 v1 = *(const f32x4*)(p + 4);
  f32x4 v2 = *(const f32x4*)(p + 16);
  f32x4 v3 = *(const f32x4*)(p + 20);
  bf16x16 r;
  #pragma unroll
  for (int i = 0; i < 4; ++i) {
    r[i]      = (__bf16)v0[i];
    r[4 + i]  = (__bf16)v1[i];
    r[8 + i]  = (__bf16)v2[i];
    r[12 + i] = (__bf16)v3[i];
  }
  return r;
}

// ============================================================================
// Kernel 1: projection GEMM.  out = Wm[M,Cc] * X[Cc,Nc] (+bias) * scale, bf16.
// TRANS=1 stores out[n*M+m] (n-major, for Q^t/K^t), else out[m*Nc+n] (for V).
// W A-frags come straight from global (L2-resident); X is double-buffered LDS.
// ============================================================================
template <bool HAS_BIAS, bool TRANS>
__global__ __launch_bounds__(256) void proj_kernel(
    const float* __restrict__ Wm, const float* __restrict__ X,
    const float* __restrict__ bias, __bf16* __restrict__ out,
    int M, float scale)
{
  __shared__ __align__(64) __bf16 Xl[2][128 * 32];  // [n][k] == B^T
  const int tx = threadIdx.x;
  const int wave = tx >> 5, lane = tx & 31;
  const int n0 = blockIdx.x * 128, m0 = blockIdx.y * 128, b = blockIdx.z;
  const float* Xb = X + (size_t)b * Cc * Nc;
  const int mrow = m0 + wave * 16 + ((lane >> 4) << 3);
  const float* wrow = Wm + (size_t)(m0 + wave * 16 + (lane & 15)) * Cc
                        + ((lane >> 4) << 3);

  const int xr = tx >> 3, xc = (tx & 7) * 16;   // X loader mapping
  f32x8 acc[8] = {};

  // stage k-step 0
  {
    const float* src = Xb + (size_t)xr * Nc + n0 + xc;
    #pragma unroll
    for (int i = 0; i < 16; ++i) Xl[0][(xc + i) * 32 + xr] = (__bf16)src[i];
  }
  __syncthreads();

  for (int ks = 0; ks < Cc / 32; ++ks) {
    const int k0 = ks * 32;
    if (ks + 1 < Cc / 32) {  // stage next k-step into the other buffer
      const float* src = Xb + (size_t)(k0 + 32 + xr) * Nc + n0 + xc;
      __bf16* dst = Xl[(ks + 1) & 1];
      #pragma unroll
      for (int i = 0; i < 16; ++i) dst[(xc + i) * 32 + xr] = (__bf16)src[i];
    }
    bf16x16 a = a_frag_from_f32(wrow + k0);
    const __bf16* Xc = Xl[ks & 1];
    #pragma unroll
    for (int nf = 0; nf < 8; ++nf) {
      bf16x16 bb = load_b_frag(Xc + nf * 16 * 32, 32, lane);
      acc[nf] = wmma_bf16(a, bb, acc[nf]);
    }
    __syncthreads();
  }

  float bv[8];
  if (HAS_BIAS) {
    f32x4 b0 = *(const f32x4*)(bias + mrow);
    f32x4 b1 = *(const f32x4*)(bias + mrow + 4);
    #pragma unroll
    for (int r = 0; r < 4; ++r) { bv[r] = b0[r]; bv[4 + r] = b1[r]; }
  } else {
    #pragma unroll
    for (int r = 0; r < 8; ++r) bv[r] = 0.0f;
  }

  __bf16* ob = out + (size_t)b * M * Nc;
  #pragma unroll
  for (int nf = 0; nf < 8; ++nf) {
    const int n = n0 + nf * 16 + (lane & 15);
    if (TRANS) {  // 8 m-values contiguous -> one packed 16B store
      bf16x8 pk;
      #pragma unroll
      for (int r = 0; r < 8; ++r) pk[r] = (__bf16)((acc[nf][r] + bv[r]) * scale);
      *(bf16x8*)(ob + (size_t)n * M + mrow) = pk;
    } else {
      #pragma unroll
      for (int r = 0; r < 8; ++r)
        ob[(size_t)(mrow + r) * Nc + n] = (__bf16)((acc[nf][r] + bv[r]) * scale);
    }
  }
}

// ============================================================================
// Kernel 2: softmax row stats (pass 1): online max/sum-exp of S[n,m]=q_n.k_m
// over all m (Q pre-scaled by 1/16).  K tiles double-buffered via async DMA.
// ============================================================================
__global__ __launch_bounds__(256) void stats_kernel(
    const __bf16* __restrict__ Qt, const __bf16* __restrict__ Kt,
    float* __restrict__ Ms, float* __restrict__ Ls)
{
  extern __shared__ __bf16 smem[];
  __bf16* Ql  = smem;                      // 128 x 256  [n][p]
  __bf16* Kb0 = smem + 128 * 256;          // 64 x 256   [m][p] buffer 0
  __bf16* Kb1 = Kb0 + 64 * 256;            // buffer 1
  const int tx = threadIdx.x, wave = tx >> 5, lane = tx & 31;
  const int n0 = blockIdx.x * 128, b = blockIdx.y;
  const __bf16* Kg = Kt + (size_t)b * Nc * Pc;
  const int kr = tx >> 2, kc = (tx & 3) * 64;

  auto issue_k = [&](int t, __bf16* dstb) {
    const __bf16* src = Kg + (size_t)(t * 64 + kr) * Pc + kc;
    __bf16* dst = dstb + kr * 256 + kc;
    #pragma unroll
    for (int i = 0; i < 8; ++i) cp_lds_b128(dst + i * 8, src + i * 8);
  };

  {  // stage Q block (64 KB) + first K tile
    int r = tx >> 1, c = (tx & 1) * 128;
    const __bf16* src = Qt + ((size_t)b * Nc + n0 + r) * Pc + c;
    __bf16* dst = Ql + r * 256 + c;
    #pragma unroll
    for (int i = 0; i < 16; ++i) cp_lds_b128(dst + i * 8, src + i * 8);
    issue_k(0, Kb0);
  }
  cp_wait();
  __syncthreads();

  bf16x16 qa[8];  // Q A-frags live in VGPRs for the whole kernel
  #pragma unroll
  for (int kk = 0; kk < 8; ++kk)
    qa[kk] = load_a_frag(Ql + wave * 16 * 256 + kk * 32, 256, lane);

  float mrun[8], srun[8];
  #pragma unroll
  for (int r = 0; r < 8; ++r) { mrun[r] = -3.0e38f; srun[r] = 0.0f; }

  for (int mt = 0; mt < MTc; ++mt) {
    const __bf16* Kcur = (mt & 1) ? Kb1 : Kb0;
    if (mt + 1 < MTc) issue_k(mt + 1, (mt & 1) ? Kb0 : Kb1);
    #pragma unroll
    for (int mf = 0; mf < 4; ++mf) {
      f32x8 s = {};
      #pragma unroll
      for (int kk = 0; kk < 8; ++kk)
        s = wmma_bf16(qa[kk], load_b_frag(Kcur + mf * 16 * 256 + kk * 32, 256, lane), s);
      #pragma unroll
      for (int r = 0; r < 8; ++r) {   // per-lane online softmax stats
        float xv = s[r];
        float mo = mrun[r];
        float mn = fmaxf(mo, xv);
        srun[r] = srun[r] * __expf(mo - mn) + __expf(xv - mn);
        mrun[r] = mn;
      }
    }
    cp_wait();
    __syncthreads();
  }

  // merge across the 16 lanes sharing each row (wave32 halves stay separate)
  #pragma unroll
  for (int off = 1; off < 16; off <<= 1) {
    #pragma unroll
    for (int r = 0; r < 8; ++r) {
      float mo = __shfl_xor(mrun[r], off, 32);
      float so = __shfl_xor(srun[r], off, 32);
      float mn = fmaxf(mrun[r], mo);
      srun[r] = srun[r] * __expf(mrun[r] - mn) + so * __expf(mo - mn);
      mrun[r] = mn;
    }
  }
  if ((lane & 15) == 0) {
    int nb = n0 + wave * 16 + ((lane >> 4) << 3);
    #pragma unroll
    for (int r = 0; r < 8; ++r) {
      Ms[(size_t)b * Nc + nb + r] = mrun[r];
      Ls[(size_t)b * Nc + nb + r] = srun[r];
    }
  }
}

// ============================================================================
// Kernel 3: output (pass 2).  Recompute S, P=exp(S-M), O += P*V^T, out=O/L+x.
// K and V tiles double-buffered via async DMA; P transposed C->A layout
// through per-wave LDS (same-wave DS ops are in-order, ISA 7.3).
// ============================================================================
__global__ __launch_bounds__(256) void attn_kernel(
    const __bf16* __restrict__ Qt, const __bf16* __restrict__ Kt,
    const __bf16* __restrict__ Vb, const float* __restrict__ Ms,
    const float* __restrict__ Ls, const float* __restrict__ xin,
    float* __restrict__ out)
{
  extern __shared__ __bf16 smem[];
  __bf16* Ql  = smem;                        // 128 x 256
  __bf16* Kb0 = smem + 32768;                // 64 x 256  [m][p]
  __bf16* Kb1 = smem + 32768 + 16384;
  __bf16* Vb0 = smem + 65536;                // 64 x 64   [c][m] == B^T
  __bf16* Vb1 = smem + 65536 + 4096;
  __bf16* Pl  = smem + 65536 + 8192;         // 8 waves x (16 x 64)
  const int tx = threadIdx.x, wave = tx >> 5, lane = tx & 31;
  const int n0 = blockIdx.x * 128, c0 = blockIdx.y * 64, b = blockIdx.z;
  const __bf16* Kg = Kt + (size_t)b * Nc * Pc;
  const __bf16* Vg = Vb + (size_t)b * Cc * Nc;
  const int kr = tx >> 2, kc = (tx & 3) * 64;
  const int vr = tx >> 2, vc = (tx & 3) * 16;

  auto issue_k = [&](int t, __bf16* dstb) {
    const __bf16* src = Kg + (size_t)(t * 64 + kr) * Pc + kc;
    __bf16* dst = dstb + kr * 256 + kc;
    #pragma unroll
    for (int i = 0; i < 8; ++i) cp_lds_b128(dst + i * 8, src + i * 8);
  };
  auto issue_v = [&](int t, __bf16* dstb) {
    const __bf16* src = Vg + (size_t)(c0 + vr) * Nc + t * 64 + vc;
    __bf16* dst = dstb + vr * 64 + vc;
    cp_lds_b128(dst, src);
    cp_lds_b128(dst + 8, src + 8);
  };

  {  // stage Q block + first K/V tiles
    int r = tx >> 1, c = (tx & 1) * 128;
    const __bf16* src = Qt + ((size_t)b * Nc + n0 + r) * Pc + c;
    __bf16* dst = Ql + r * 256 + c;
    #pragma unroll
    for (int i = 0; i < 16; ++i) cp_lds_b128(dst + i * 8, src + i * 8);
    issue_k(0, Kb0);
    issue_v(0, Vb0);
  }
  cp_wait();
  __syncthreads();

  bf16x16 qa[8];
  #pragma unroll
  for (int kk = 0; kk < 8; ++kk)
    qa[kk] = load_a_frag(Ql + wave * 16 * 256 + kk * 32, 256, lane);

  float Mr[8], iLr[8];
  const int nbase = n0 + wave * 16 + ((lane >> 4) << 3);
  #pragma unroll
  for (int r = 0; r < 8; ++r) {
    Mr[r]  = Ms[(size_t)b * Nc + nbase + r];
    iLr[r] = 1.0f / Ls[(size_t)b * Nc + nbase + r];
  }

  f32x8 oacc[4] = {};
  __bf16* Pw = Pl + wave * 16 * 64;

  for (int mt = 0; mt < MTc; ++mt) {
    const __bf16* Kcur = (mt & 1) ? Kb1 : Kb0;
    const __bf16* Vcur = (mt & 1) ? Vb1 : Vb0;
    if (mt + 1 < MTc) {
      issue_k(mt + 1, (mt & 1) ? Kb0 : Kb1);
      issue_v(mt + 1, (mt & 1) ? Vb0 : Vb1);
    }

    // S tiles -> P (bf16) into per-wave LDS region, A-layout source [n][m]
    #pragma unroll
    for (int mf = 0; mf < 4; ++mf) {
      f32x8 s = {};
      #pragma unroll
      for (int kk = 0; kk < 8; ++kk)
        s = wmma_bf16(qa[kk], load_b_frag(Kcur + mf * 16 * 256 + kk * 32, 256, lane), s);
      #pragma unroll
      for (int r = 0; r < 8; ++r) {
        float p = __expf(s[r] - Mr[r]);
        Pw[(r + ((lane >> 4) << 3)) * 64 + mf * 16 + (lane & 15)] = (__bf16)p;
      }
    }
    asm volatile("" ::: "memory");  // keep P stores before P reloads (same-wave DS in-order)

    // O += P[16x64] * V^T[64x64]  (2 k-steps of 32)
    #pragma unroll
    for (int k2 = 0; k2 < 2; ++k2) {
      bf16x16 pa = load_a_frag(Pw + k2 * 32, 64, lane);
      #pragma unroll
      for (int cf = 0; cf < 4; ++cf) {
        bf16x16 vv = load_b_frag(Vcur + cf * 16 * 64 + k2 * 32, 64, lane);
        oacc[cf] = wmma_bf16(pa, vv, oacc[cf]);
      }
    }
    cp_wait();
    __syncthreads();
  }

  // epilogue: out[b][c][n] = O/L + x   (8 n-values contiguous -> f32x4 pairs)
  #pragma unroll
  for (int cf = 0; cf < 4; ++cf) {
    const int c = c0 + cf * 16 + (lane & 15);
    const size_t idx0 = ((size_t)b * Cc + c) * (size_t)Nc + nbase;
    f32x4 x0 = *(const f32x4*)(xin + idx0);
    f32x4 x1 = *(const f32x4*)(xin + idx0 + 4);
    f32x4 o0, o1;
    #pragma unroll
    for (int r = 0; r < 4; ++r) {
      o0[r] = oacc[cf][r] * iLr[r] + x0[r];
      o1[r] = oacc[cf][4 + r] * iLr[4 + r] + x1[r];
    }
    *(f32x4*)(out + idx0) = o0;
    *(f32x4*)(out + idx0 + 4) = o1;
  }
}

// ============================================================================
extern "C" void kernel_launch(void* const* d_in, const int* in_sizes, int n_in,
                              void* d_out, int out_size, void* d_ws, size_t ws_size,
                              hipStream_t stream) {
  (void)in_sizes; (void)n_in; (void)out_size; (void)ws_size;
  const float* x  = (const float*)d_in[0];
  const float* Wv = (const float*)d_in[1];
  const float* Wq = (const float*)d_in[2];
  const float* bq = (const float*)d_in[3];
  const float* Wk = (const float*)d_in[4];
  const float* bk = (const float*)d_in[5];
  float* out = (float*)d_out;

  char* ws = (char*)d_ws;
  __bf16* Qt = (__bf16*)ws;  ws += (size_t)Bc * Nc * Pc * sizeof(__bf16);  // [B][N][P], pre-scaled
  __bf16* Kt = (__bf16*)ws;  ws += (size_t)Bc * Nc * Pc * sizeof(__bf16);  // [B][N][P]
  __bf16* Vv = (__bf16*)ws;  ws += (size_t)Bc * Cc * Nc * sizeof(__bf16);  // [B][C][N]
  float*  Ms = (float*)ws;   ws += (size_t)Bc * Nc * sizeof(float);
  float*  Ls = (float*)ws;   ws += (size_t)Bc * Nc * sizeof(float);

  dim3 blk(256);
  proj_kernel<false, false><<<dim3(Nc / 128, Cc / 128, Bc), blk, 0, stream>>>(
      Wv, x, nullptr, Vv, Cc, 1.0f);
  proj_kernel<true, true><<<dim3(Nc / 128, Pc / 128, Bc), blk, 0, stream>>>(
      Wq, x, bq, Qt, Pc, QSCALE);
  proj_kernel<true, true><<<dim3(Nc / 128, Pc / 128, Bc), blk, 0, stream>>>(
      Wk, x, bk, Kt, Pc, 1.0f);

  size_t smem1 = (size_t)(128 * 256 + 2 * 64 * 256) * sizeof(__bf16);  // 128 KB
  stats_kernel<<<dim3(Nc / 128, Bc), blk, smem1, stream>>>(Qt, Kt, Ms, Ls);

  size_t smem2 = (size_t)(128 * 256 + 2 * 64 * 256 + 2 * 64 * 64 + 8 * 16 * 64)
                 * sizeof(__bf16);                                      // 160 KB
  attn_kernel<<<dim3(Nc / 128, Cc / 64, Bc), blk, smem2, stream>>>(
      Qt, Kt, Vv, Ms, Ls, x, out);
}
